// MultiHeadAttention_18279380812583
// MI455X (gfx1250) — matrix operand order repaired and checked by reading, not compiled, not run on hardware
//
#include <hip/hip_runtime.h>

// Problem constants (match reference): B=2, S=2048, D=1024, H=16, DK=64
constexpr int kB = 2, kS = 2048, kD = 1024, kH = 16, kDK = 64;

typedef __attribute__((ext_vector_type(16))) _Float16 v16h;
typedef __attribute__((ext_vector_type(8)))  _Float16 v8h;
typedef __attribute__((ext_vector_type(8)))  float    v8f;
typedef __attribute__((ext_vector_type(4)))  float    v4f;
typedef __attribute__((ext_vector_type(4)))  unsigned int v4u;

static __device__ __forceinline__ v8f vzero8() {
  v8f z = {0.f, 0.f, 0.f, 0.f, 0.f, 0.f, 0.f, 0.f};
  return z;
}

// D = A(16x32 f16) * B(32x16 f16) + C(16x16 f32)
static __device__ __forceinline__ v8f wmma_f16(v16h a, v16h b, v8f c) {
  return __builtin_amdgcn_wmma_f32_16x16x32_f16(
      /*neg_a=*/false, a, /*neg_b=*/false, b,
      /*c_mod=*/(short)0, c, /*reuse_a=*/false, /*reuse_b=*/false);
}

// A-fragment (16x32 f16, MxK) from row-major f16 memory (global or LDS):
// lane holds row M=lane%16; halves [0..15] = K (hi*16 + 0..15),
// relative to p = row_base + k0 + hi*16.
static __device__ __forceinline__ v16h load_a16(const _Float16* p) {
  v16h a;
  ((v4u*)&a)[0] = *(const v4u*)(p);        // 8 contiguous halves
  ((v4u*)&a)[1] = *(const v4u*)(p + 8);    // next 8 contiguous halves
  return a;
}

// B-fragment (32x16 f16, KxN): lane holds column N=lane%16, 16 contiguous K
// values (hi selects K 0..15 vs 16..31); p = col_base + k0 + hi*16.
static __device__ __forceinline__ v16h load_b16(const _Float16* p) {
  return *(const v16h*)(p);  // 32 contiguous bytes -> 2x b128
}

// ---------------------------------------------------------------------------
// Kernel 0: bulk fp32 -> f16 conversion (one pass over X and W operands).
// 8 elements per thread: 32B in (2x b128), 16B out (1x b128).
// ---------------------------------------------------------------------------
__global__ __launch_bounds__(256) void cvt_f32_f16_kernel(
    const float* __restrict__ src, _Float16* __restrict__ dst, int n) {
  const int idx = (blockIdx.x * 256 + threadIdx.x) * 8;
  if (idx >= n) return;
  const v4f x0 = *(const v4f*)(src + idx);
  const v4f x1 = *(const v4f*)(src + idx + 4);
  v8h y;
#pragma unroll
  for (int e = 0; e < 4; ++e) {
    y[e]     = (_Float16)x0[e];
    y[4 + e] = (_Float16)x1[e];
  }
  *(v8h*)(dst + idx) = y;
}

// ---------------------------------------------------------------------------
// Kernel 1: fused QKV projection.  y = x @ W^T + b on pre-converted f16
// operands, fp32 WMMA accumulation.  Q,K written f16 [b,h,s,dk]; V written
// TRANSPOSED f16 [b,h,dk,s] so attention PV B-fragments are contiguous.
// Wave tile: 64x64 (4x4 WMMAs per 32-wide k-step; inner loop = loads+WMMA).
// ---------------------------------------------------------------------------
__global__ __launch_bounds__(256) void qkv_proj_kernel(
    const _Float16* __restrict__ xq16, const _Float16* __restrict__ xk16,
    const _Float16* __restrict__ xv16,
    const _Float16* __restrict__ Wq16, const _Float16* __restrict__ Wk16,
    const _Float16* __restrict__ Wv16,
    const float* __restrict__ bq, const float* __restrict__ bk,
    const float* __restrict__ bv,
    _Float16* __restrict__ Qh, _Float16* __restrict__ Kh,
    _Float16* __restrict__ Vt) {
  const int lane = threadIdx.x & 31;
  const int lo = lane & 15, hi = lane >> 4;
  const int wid = blockIdx.x * 8 + (threadIdx.x >> 5);
  constexpr int MT = (kB * kS) / 64;  // 64
  constexpr int NT = kD / 64;         // 16
  const int proj = wid / (MT * NT);
  const int rem  = wid % (MT * NT);
  const int m0 = (rem / NT) * 64;
  const int n0 = (rem % NT) * 64;

  const _Float16* X  = (proj == 0) ? xq16 : (proj == 1) ? xk16 : xv16;
  const _Float16* W  = (proj == 0) ? Wq16 : (proj == 1) ? Wk16 : Wv16;
  const float* bias  = (proj == 0) ? bq : (proj == 1) ? bk : bv;

  v8f acc[4][4];
#pragma unroll
  for (int mi = 0; mi < 4; ++mi)
#pragma unroll
    for (int nj = 0; nj < 4; ++nj) acc[mi][nj] = vzero8();

  for (int k0 = 0; k0 < kD; k0 += 32) {
    v16h a[4], bf[4];
#pragma unroll
    for (int mi = 0; mi < 4; ++mi)
      a[mi] = load_a16(X + (size_t)(m0 + mi * 16 + lo) * kD + k0 + hi * 16);
#pragma unroll
    for (int nj = 0; nj < 4; ++nj)
      bf[nj] = load_b16(W + (size_t)(n0 + nj * 16 + lo) * kD + k0 + hi * 16);
#pragma unroll
    for (int mi = 0; mi < 4; ++mi)
#pragma unroll
      for (int nj = 0; nj < 4; ++nj)
        acc[mi][nj] = wmma_f16(a[mi], bf[nj], acc[mi][nj]);
  }

  // Epilogue: +bias (fp32), convert to f16, scatter to head-split layouts.
#pragma unroll
  for (int mi = 0; mi < 4; ++mi) {
#pragma unroll
    for (int nj = 0; nj < 4; ++nj) {
#pragma unroll
      for (int i = 0; i < 8; ++i) {
        const int row = m0 + mi * 16 + i + 8 * hi;  // b*S + s
        const int col = n0 + nj * 16 + lo;          // h*DK + dk
        const float val = acc[mi][nj][i] + bias[col];
        const int bb = row / kS, ss = row % kS;
        const int hh = col / kDK, dk = col % kDK;
        if (proj == 2) {
          Vt[(((size_t)bb * kH + hh) * kDK + dk) * kS + ss] = (_Float16)val;
        } else {
          _Float16* dst = (proj == 0) ? Qh : Kh;
          dst[(((size_t)bb * kH + hh) * kS + ss) * kDK + dk] = (_Float16)val;
        }
      }
    }
  }
}

// ---------------------------------------------------------------------------
// Kernel 2: flash attention.  One wave owns 16 query rows of one (b,h);
// iterates over key blocks of 32 with online softmax.  Scores and output are
// fp32 WMMA accumulators; P re-layout via a wave-private 1 KB LDS tile.
// All operands f16 and fully L2-resident (Q+K+Vt = 24 MB << 192 MB L2).
// ---------------------------------------------------------------------------
__global__ __launch_bounds__(256) void flash_attn_kernel(
    const _Float16* __restrict__ Qh, const _Float16* __restrict__ Kh,
    const _Float16* __restrict__ Vt, const int* __restrict__ mask,
    _Float16* __restrict__ attn) {
  __shared__ __align__(32) _Float16 pbuf[8][16 * 32];
  const int lane = threadIdx.x & 31;
  const int lo = lane & 15, hi = lane >> 4;
  const int wib = threadIdx.x >> 5;
  const int wid = blockIdx.x * 8 + wib;
  const int qb = wid % (kS / 16);
  const int bh = wid / (kS / 16);
  const int b = bh / kH;
  const int h = bh % kH;
  const int q0 = qb * 16;

  const _Float16* Qb = Qh + (size_t)bh * kS * kDK;
  const _Float16* Kb = Kh + (size_t)bh * kS * kDK;
  const _Float16* Vb = Vt + (size_t)bh * kDK * kS;
  const int* mbase = mask + (size_t)b * kS * kS;

  // Q fragments resident for the whole loop (DK=64 -> 2 k-steps of 32).
  v16h aq[2];
#pragma unroll
  for (int kk = 0; kk < 2; ++kk)
    aq[kk] = load_a16(Qb + (size_t)(q0 + lo) * kDK + kk * 32 + hi * 16);

  v8f acc[4];
  float m[8], l[8];
#pragma unroll
  for (int j = 0; j < 4; ++j) acc[j] = vzero8();
#pragma unroll
  for (int i = 0; i < 8; ++i) { m[i] = -3.0e38f; l[i] = 0.f; }

  _Float16* pl = pbuf[wib];
  const float sc  = 0.125f;                // 1/sqrt(DK)
  const float L2E = 1.4426950408889634f;   // log2(e)

  for (int t0 = 0; t0 < kS; t0 += 32) {
    if (t0 + 32 < kS) {  // prefetch next K/V block (L2-resident stream)
      __builtin_prefetch(Kb + (size_t)(t0 + 32 + lo) * kDK, 0, 3);
      __builtin_prefetch(Vb + (size_t)lo * kS + t0 + 32, 0, 3);
    }
    // scores S[16x32] = Q @ K^T  (two 16-wide N tiles)
    v8f s0 = vzero8(), s1 = vzero8();
#pragma unroll
    for (int kk = 0; kk < 2; ++kk) {
      v16h bk0 = load_b16(Kb + (size_t)(t0 + lo) * kDK + kk * 32 + hi * 16);
      v16h bk1 = load_b16(Kb + (size_t)(t0 + 16 + lo) * kDK + kk * 32 + hi * 16);
      s0 = wmma_f16(aq[kk], bk0, s0);
      s1 = wmma_f16(aq[kk], bk1, s1);
    }
    // masked scale + online softmax per row (row = i + 8*hi, col = lane%16)
#pragma unroll
    for (int i = 0; i < 8; ++i) {
      const int qrow = q0 + i + 8 * hi;
      const int* mr = mbase + (size_t)qrow * kS + t0;
      float v0 = (mr[lo] == 0)      ? -1e9f : s0[i] * sc;
      float v1 = (mr[16 + lo] == 0) ? -1e9f : s1[i] * sc;
      float r = fmaxf(v0, v1);
#pragma unroll
      for (int off = 1; off < 16; off <<= 1)
        r = fmaxf(r, __shfl_xor(r, off, 32));
      const float mn = fmaxf(m[i], r);
      const float alpha = __builtin_amdgcn_exp2f((m[i] - mn) * L2E);
      const float p0 = __builtin_amdgcn_exp2f((v0 - mn) * L2E);
      const float p1 = __builtin_amdgcn_exp2f((v1 - mn) * L2E);
      m[i] = mn;
      float ps = p0 + p1;
#pragma unroll
      for (int off = 1; off < 16; off <<= 1)
        ps += __shfl_xor(ps, off, 32);
      l[i] = l[i] * alpha + ps;
#pragma unroll
      for (int j = 0; j < 4; ++j) acc[j][i] *= alpha;
      // stage P as f16 into wave-private LDS tile [16 rows][32 keys]
      pl[(i + 8 * hi) * 32 + lo]      = (_Float16)p0;
      pl[(i + 8 * hi) * 32 + 16 + lo] = (_Float16)p1;
    }
    // re-load P as an A-fragment (DS ops are in-order within a wave)
    v16h ap = load_a16(pl + lo * 32 + hi * 16);
    // acc[16x64] += P @ V   (V stored transposed: contiguous in t per dk row)
#pragma unroll
    for (int j = 0; j < 4; ++j) {
      v16h bv = load_b16(Vb + (size_t)(j * 16 + lo) * kS + t0 + hi * 16);
      acc[j] = wmma_f16(ap, bv, acc[j]);
    }
  }

  // Normalize and merge heads: attn[b, s, h*DK + dk] (f16)
#pragma unroll
  for (int i = 0; i < 8; ++i) {
    const int qrow = q0 + i + 8 * hi;
    const float rl = __builtin_amdgcn_rcpf(l[i]);
#pragma unroll
    for (int j = 0; j < 4; ++j) {
      const float o = acc[j][i] * rl;
      attn[((size_t)b * kS + qrow) * kD + h * kDK + j * 16 + lo] = (_Float16)o;
    }
  }
}

// ---------------------------------------------------------------------------
// Kernel 3: output projection.  out = attn(f16) @ Wo16^T + bo, fp32 result.
// ---------------------------------------------------------------------------
__global__ __launch_bounds__(256) void out_proj_kernel(
    const _Float16* __restrict__ A, const _Float16* __restrict__ Wo16,
    const float* __restrict__ bo, float* __restrict__ out) {
  const int lane = threadIdx.x & 31;
  const int lo = lane & 15, hi = lane >> 4;
  const int wid = blockIdx.x * 8 + (threadIdx.x >> 5);
  constexpr int NT = kD / 64;  // 16
  const int m0 = (wid / NT) * 64;
  const int n0 = (wid % NT) * 64;

  v8f acc[4][4];
#pragma unroll
  for (int mi = 0; mi < 4; ++mi)
#pragma unroll
    for (int nj = 0; nj < 4; ++nj) acc[mi][nj] = vzero8();

  for (int k0 = 0; k0 < kD; k0 += 32) {
    v16h a[4], bf[4];
#pragma unroll
    for (int mi = 0; mi < 4; ++mi)
      a[mi] = load_a16(A + (size_t)(m0 + mi * 16 + lo) * kD + k0 + hi * 16);
#pragma unroll
    for (int nj = 0; nj < 4; ++nj)
      bf[nj] = load_b16(Wo16 + (size_t)(n0 + nj * 16 + lo) * kD + k0 + hi * 16);
#pragma unroll
    for (int mi = 0; mi < 4; ++mi)
#pragma unroll
      for (int nj = 0; nj < 4; ++nj)
        acc[mi][nj] = wmma_f16(a[mi], bf[nj], acc[mi][nj]);
  }

#pragma unroll
  for (int mi = 0; mi < 4; ++mi)
#pragma unroll
    for (int nj = 0; nj < 4; ++nj)
#pragma unroll
      for (int i = 0; i < 8; ++i) {
        const int row = m0 + mi * 16 + i + 8 * hi;
        const int col = n0 + nj * 16 + lo;
        out[(size_t)row * kD + col] = acc[mi][nj][i] + bo[col];
      }
}

// ---------------------------------------------------------------------------
extern "C" void kernel_launch(void* const* d_in, const int* in_sizes, int n_in,
                              void* d_out, int out_size, void* d_ws,
                              size_t ws_size, hipStream_t stream) {
  const float* q  = (const float*)d_in[0];
  const float* k  = (const float*)d_in[1];
  const float* v  = (const float*)d_in[2];
  const int* mask = (const int*)d_in[3];
  const float* Wq = (const float*)d_in[4];
  const float* bq = (const float*)d_in[5];
  const float* Wk = (const float*)d_in[6];
  const float* bk = (const float*)d_in[7];
  const float* Wv = (const float*)d_in[8];
  const float* bv = (const float*)d_in[9];
  const float* Wo = (const float*)d_in[10];
  const float* bo = (const float*)d_in[11];
  float* out = (float*)d_out;

  // Workspace layout (all f16):
  //   Qh | Kh | Vt | At            : 4 * E  halves (32 MB)
  //   Xq16 | Xk16 | Xv16           : 3 * E  halves (24 MB)
  //   Wq16 | Wk16 | Wv16 | Wo16    : 4 * WE halves ( 8 MB)
  const size_t E  = (size_t)kB * kS * kD;  // 4,194,304
  const size_t WE = (size_t)kD * kD;       // 1,048,576
  _Float16* Qh   = (_Float16*)d_ws;
  _Float16* Kh   = Qh + E;
  _Float16* Vt   = Kh + E;
  _Float16* At   = Vt + E;
  _Float16* Xq16 = At + E;
  _Float16* Xk16 = Xq16 + E;
  _Float16* Xv16 = Xk16 + E;
  _Float16* Wq16 = Xv16 + E;
  _Float16* Wk16 = Wq16 + WE;
  _Float16* Wv16 = Wk16 + WE;
  _Float16* Wo16 = Wv16 + WE;

  // One-pass fp32 -> f16 conversion of all GEMM operands.
  const int XB = (int)(E / (256 * 8));   // 2048 blocks
  const int WB = (int)(WE / (256 * 8));  // 512 blocks
  cvt_f32_f16_kernel<<<XB, 256, 0, stream>>>(q, Xq16, (int)E);
  cvt_f32_f16_kernel<<<XB, 256, 0, stream>>>(k, Xk16, (int)E);
  cvt_f32_f16_kernel<<<XB, 256, 0, stream>>>(v, Xv16, (int)E);
  cvt_f32_f16_kernel<<<WB, 256, 0, stream>>>(Wq, Wq16, (int)WE);
  cvt_f32_f16_kernel<<<WB, 256, 0, stream>>>(Wk, Wk16, (int)WE);
  cvt_f32_f16_kernel<<<WB, 256, 0, stream>>>(Wv, Wv16, (int)WE);
  cvt_f32_f16_kernel<<<WB, 256, 0, stream>>>(Wo, Wo16, (int)WE);

  // 3 projections * 64 M-tiles * 16 N-tiles = 3072 waves / 8 per block
  qkv_proj_kernel<<<384, 256, 0, stream>>>(Xq16, Xk16, Xv16, Wq16, Wk16, Wv16,
                                           bq, bk, bv, Qh, Kh, Vt);
  // B*H*(S/16) = 4096 waves / 8 per block
  flash_attn_kernel<<<512, 256, 0, stream>>>(Qh, Kh, Vt, mask, At);
  // 64 M-tiles * 16 N-tiles = 1024 waves / 8 per block
  out_proj_kernel<<<128, 256, 0, stream>>>(At, Wo16, bo, out);
}
